// Point_M2AE_35278861369741
// MI455X (gfx1250) — matrix operand order repaired and checked by reading, not compiled, not run on hardware
//
#include <hip/hip_runtime.h>
#include <hip/hip_bf16.h>
#include <stdint.h>

// ---------------- constants ----------------
#define B_     16
#define G_     2048
#define C_     384
#define H_     6
#define DEPTH_ 6
#define V_     820
#define VPAD   832          // 52 * 16
#define M_     (B_ * V_)    // 13120 = 820 * 16  (divisible by 16!)
#define DH     64
#define MW     32           // mask words allocated per row (26 used)
#define NEGBIG (-1.0e30f)

typedef __attribute__((ext_vector_type(16))) _Float16 v16h;
typedef __attribute__((ext_vector_type(8)))  float    v8f;

// ---------------- WMMA fragment helpers (wave32, 16x16x32 f16) ----------------
// A 16x32 f16: lane m = lane&15; group g = lane>>4;
//   VGPR v: K = g*16 + v*2 (pair)
__device__ __forceinline__ v16h load_a16(const _Float16* base, int ld, int row0, int k0) {
  int lane = threadIdx.x & 31;
  int g = lane >> 4;
  const uint32_t* p = (const uint32_t*)(base + (size_t)(row0 + (lane & 15)) * ld + k0);
  union { v16h v; uint32_t u[8]; } r;
#pragma unroll
  for (int i = 0; i < 8; ++i) {
    r.u[i] = p[g * 8 + i];
  }
  return r.v;
}

// B 32x16 f16 from row-major Wt[N][K]: lane n = n0 + (lane&15);
//   lanes 0-15 cover K=0..15, lanes 16-31 cover K=16..31; VGPR v: K pair 2v,2v+1
__device__ __forceinline__ v16h load_b16(const _Float16* base, int ld, int n0, int k0) {
  int lane = threadIdx.x & 31;
  int g = lane >> 4;
  const uint32_t* p = (const uint32_t*)(base + (size_t)(n0 + (lane & 15)) * ld + k0 + g * 16);
  union { v16h v; uint32_t u[8]; } r;
#pragma unroll
  for (int i = 0; i < 8; ++i) r.u[i] = p[i];
  return r.v;
}

__device__ __forceinline__ v8f wmma_f16(v16h a, v16h b, v8f c) {
  return __builtin_amdgcn_wmma_f32_16x16x32_f16(false, a, false, b, (short)0, c, false, false);
}

__device__ __forceinline__ float gelu_f(float x) {
  float x3 = x * x * x;
  return 0.5f * x * (1.0f + tanhf(0.7978845608028654f * (x + 0.044715f * x3)));
}

// ---------------- small prep kernels ----------------
__global__ void order_kernel(const unsigned char* mask, int* order) {
  int b = threadIdx.x;
  if (b < B_) {
    int cnt = 0;
    for (int gg = 0; gg < G_; ++gg)
      if (mask[b * G_ + gg] == 0) { if (cnt < V_) order[b * V_ + cnt] = gg; cnt++; }
  }
}

__global__ void gather_tokens(const float* tokens, const int* order, float* xb) {
  int i = blockIdx.x * blockDim.x + threadIdx.x;
  if (i >= M_ * C_) return;
  int c = i % C_;
  int m = i / C_;
  int b = m / V_;
  int gg = order[m];
  xb[i] = tokens[((size_t)b * G_ + gg) * C_ + c];
}

__global__ void gather_ctr(const float* centers, const int* order, float* ctr) {
  int i = blockIdx.x * blockDim.x + threadIdx.x;
  if (i >= M_ * 3) return;
  int j = i % 3;
  int m = i / 3;
  int b = m / V_;
  ctr[i] = centers[((size_t)b * G_ + order[m]) * 3 + j];
}

// blocked bit = (dist >= 0.4)  <=>  d2 >= 0.16 ; keys >= V blocked
__global__ void maskbits_kernel(const float* ctr, uint32_t* mb) {
  int i = blockIdx.x * blockDim.x + threadIdx.x;
  if (i >= B_ * V_ * 26) return;
  int w = i % 26;
  int q = (i / 26) % V_;
  int b = i / (26 * V_);
  const float* cq = ctr + ((size_t)b * V_ + q) * 3;
  float qx = cq[0], qy = cq[1], qz = cq[2];
  uint32_t bits = 0;
  for (int j = 0; j < 32; ++j) {
    int k = w * 32 + j;
    int blocked = 1;
    if (k < V_) {
      const float* ck = ctr + ((size_t)b * V_ + k) * 3;
      float dx = qx - ck[0], dy = qy - ck[1], dz = qz - ck[2];
      blocked = (dx * dx + dy * dy + dz * dz) >= 0.16f;
    }
    bits |= ((uint32_t)blocked) << j;
  }
  mb[((size_t)b * V_ + q) * MW + w] = bits;
}

// W[K][N] f32 -> Wt[N][K] f16
__global__ void convT_kernel(const float* W, _Float16* Wt, int K, int N, int total) {
  int i = blockIdx.x * blockDim.x + threadIdx.x;
  if (i >= total) return;
  int k = i % K;
  int n = i / K;
  Wt[i] = (_Float16)W[(size_t)k * N + n];
}

// pos1 = gelu(ctr @ w1 + b1)  (K=3, done scalar)
__global__ void posmlp1_kernel(const float* ctr, const float* w1, const float* b1, _Float16* out) {
  int i = blockIdx.x * blockDim.x + threadIdx.x;
  if (i >= M_ * C_) return;
  int c = i % C_;
  int m = i / C_;
  const float* cm = ctr + (size_t)m * 3;
  float s = b1[c] + cm[0] * w1[c] + cm[1] * w1[C_ + c] + cm[2] * w1[2 * C_ + c];
  out[i] = (_Float16)gelu_f(s);
}

// zero the key-padding columns of transposed V so p=0 never meets NaN in WMMA
__global__ void padv_kernel(_Float16* vT) {
  const int PAD = VPAD - V_;
  int i = blockIdx.x * blockDim.x + threadIdx.x;
  if (i >= B_ * H_ * DH * PAD) return;
  int p = i % PAD;
  int rest = i / PAD;
  int d = rest % DH;
  int bh = rest / DH;
  vT[((size_t)bh * DH + d) * VPAD + V_ + p] = (_Float16)0.f;
}

// ---------------- LayerNorm (block per token row) ----------------
// MODE 0: v = in + pos, write hout(f32) and out16 ; MODE 1: out16 only ; MODE 2: out32 only
template <int MODE>
__global__ void ln_kernel(const float* in, const float* pos, const float* w, const float* bsc,
                          float* hout, _Float16* out16, float* out32) {
  int m = blockIdx.x;
  int tid = threadIdx.x; // 128
  __shared__ float red[128];
  float vals[3];
#pragma unroll
  for (int i = 0; i < 3; ++i) {
    int c = tid + i * 128;
    float v = in[(size_t)m * C_ + c];
    if (MODE == 0) { v += pos[(size_t)m * C_ + c]; hout[(size_t)m * C_ + c] = v; }
    vals[i] = v;
  }
  red[tid] = vals[0] + vals[1] + vals[2];
  __syncthreads();
  for (int off = 64; off > 0; off >>= 1) {
    if (tid < off) red[tid] += red[tid + off];
    __syncthreads();
  }
  float mean = red[0] * (1.0f / C_);
  __syncthreads();
  float s2 = 0.f;
#pragma unroll
  for (int i = 0; i < 3; ++i) { float d = vals[i] - mean; s2 += d * d; }
  red[tid] = s2;
  __syncthreads();
  for (int off = 64; off > 0; off >>= 1) {
    if (tid < off) red[tid] += red[tid + off];
    __syncthreads();
  }
  float rstd = rsqrtf(red[0] * (1.0f / C_) + 1e-5f);
#pragma unroll
  for (int i = 0; i < 3; ++i) {
    int c = tid + i * 128;
    float o = (vals[i] - mean) * rstd * w[c] + bsc[c];
    if (MODE <= 1) out16[(size_t)m * C_ + c] = (_Float16)o;
    if (MODE == 2) out32[(size_t)m * C_ + c] = o;
  }
}

// ---------------- generic WMMA GEMM: D[M,N] = A[M,K](f16) x Wt[N,K](f16) + bias ----------------
// EPI 0: write f32                      (pos MLP layer 2)
// EPI 1: write f32 = res + val          (proj / fc2 residual)
// EPI 2: write f16 = gelu(val)          (fc1)
// EPI 3: qkv scatter to q16/k16/vT(f16)
template <int EPI>
__global__ void gemm_wmma(const _Float16* A, const _Float16* Wt, const float* bias,
                          int K, int N,
                          float* out32, const float* res, _Float16* out16,
                          _Float16* q16, _Float16* k16, _Float16* vT) {
  int m0 = blockIdx.x * 16;
  int n0 = blockIdx.y * 64;
  int lane = threadIdx.x & 31;
  int g = lane >> 4;
  int nn = lane & 15;
  v8f acc[4];
#pragma unroll
  for (int c = 0; c < 4; ++c) acc[c] = (v8f){0.f, 0.f, 0.f, 0.f, 0.f, 0.f, 0.f, 0.f};

  for (int kb = 0; kb < K; kb += 32) {
    __builtin_prefetch(A + (size_t)(m0 + nn) * K + kb + 64, 0, 1); // global_prefetch_b8
    v16h a  = load_a16(A, K, m0, kb);
    v16h b0 = load_b16(Wt, K, n0, kb);
    v16h b1 = load_b16(Wt, K, n0 + 16, kb);
    v16h b2 = load_b16(Wt, K, n0 + 32, kb);
    v16h b3 = load_b16(Wt, K, n0 + 48, kb);
    acc[0] = wmma_f16(a, b0, acc[0]);
    acc[1] = wmma_f16(a, b1, acc[1]);
    acc[2] = wmma_f16(a, b2, acc[2]);
    acc[3] = wmma_f16(a, b3, acc[3]);
  }

#pragma unroll
  for (int c = 0; c < 4; ++c) {
#pragma unroll
    for (int r = 0; r < 8; ++r) {
      int row = m0 + r + 8 * g;
      int col = n0 + c * 16 + nn;
      float val = acc[c][r] + bias[col];
      if (EPI == 0) {
        out32[(size_t)row * N + col] = val;
      } else if (EPI == 1) {
        out32[(size_t)row * N + col] = res[(size_t)row * N + col] + val;
      } else if (EPI == 2) {
        out16[(size_t)row * N + col] = (_Float16)gelu_f(val);
      } else {
        int i3 = col / (H_ * DH);     // 0=q 1=k 2=v
        int hh = (col >> 6) % H_;
        int d  = col & 63;
        int b  = row / V_;
        int vt = row % V_;
        int bh = b * H_ + hh;
        if (i3 == 0)      q16[((size_t)bh * VPAD + vt) * DH + d] = (_Float16)val;
        else if (i3 == 1) k16[((size_t)bh * VPAD + vt) * DH + d] = (_Float16)val;
        else              vT [((size_t)bh * DH + d) * VPAD + vt] = (_Float16)val;
      }
    }
  }
}

// ---------------- flash-style masked attention: one wave per (b,h,q-tile16) ----------------
__global__ void attn_kernel(const _Float16* q16, const _Float16* k16, const _Float16* vT,
                            const uint32_t* maskbits, _Float16* o16) {
  const int QT = VPAD / 16; // 52
  int qt = blockIdx.x % QT;
  int bh = blockIdx.x / QT;
  int b = bh / H_;
  int h = bh % H_;
  int lane = threadIdx.x & 31;
  int g = lane >> 4;
  int nn = lane & 15;
  __shared__ _Float16 P[16][32];

  const _Float16* Qb = q16 + (size_t)bh * VPAD * DH;
  const _Float16* Kb = k16 + (size_t)bh * VPAD * DH;
  const _Float16* Vb = vT + (size_t)bh * DH * VPAD;

  v16h aq0 = load_a16(Qb, DH, qt * 16, 0);
  v16h aq1 = load_a16(Qb, DH, qt * 16, 32);

  float mrow[8], lrow[8];
  v8f o[4];
  const v8f zacc = (v8f){0.f, 0.f, 0.f, 0.f, 0.f, 0.f, 0.f, 0.f};
#pragma unroll
  for (int r = 0; r < 8; ++r) { mrow[r] = NEGBIG; lrow[r] = 0.f; }
#pragma unroll
  for (int c = 0; c < 4; ++c) o[c] = zacc;

  for (int kt = 0; kt < VPAD; kt += 32) {
    // scores S[16 q, 32 k] = Q Kt  (two 16-col fragments, d split 0..31 / 32..63)
    v8f s0 = wmma_f16(aq0, load_b16(Kb, DH, kt, 0), zacc);
    s0     = wmma_f16(aq1, load_b16(Kb, DH, kt, 32), s0);
    v8f s1 = wmma_f16(aq0, load_b16(Kb, DH, kt + 16, 0), zacc);
    s1     = wmma_f16(aq1, load_b16(Kb, DH, kt + 16, 32), s1);

#pragma unroll
    for (int r = 0; r < 8; ++r) {
      int qrow = qt * 16 + r + 8 * g;
      uint32_t wd = maskbits[((size_t)b * V_ + qrow) * MW + (kt >> 5)];
      int key0 = kt + nn, key1 = kt + 16 + nn;
      bool blk0 = (qrow >= V_) || (key0 >= V_) || (((wd >> nn) & 1u) != 0);
      bool blk1 = (qrow >= V_) || (key1 >= V_) || (((wd >> (16 + nn)) & 1u) != 0);
      float sv0 = blk0 ? NEGBIG : s0[r] * 0.125f;
      float sv1 = blk1 ? NEGBIG : s1[r] * 0.125f;
      float mx = fmaxf(sv0, sv1);
      mx = fmaxf(mx, __shfl_xor(mx, 1));
      mx = fmaxf(mx, __shfl_xor(mx, 2));
      mx = fmaxf(mx, __shfl_xor(mx, 4));
      mx = fmaxf(mx, __shfl_xor(mx, 8));
      float mnew = fmaxf(mrow[r], mx);
      float corr = __expf(mrow[r] - mnew);
      float p0 = blk0 ? 0.f : __expf(sv0 - mnew);
      float p1 = blk1 ? 0.f : __expf(sv1 - mnew);
      float ps = p0 + p1;
      ps += __shfl_xor(ps, 1);
      ps += __shfl_xor(ps, 2);
      ps += __shfl_xor(ps, 4);
      ps += __shfl_xor(ps, 8);
      lrow[r] = lrow[r] * corr + ps;
      mrow[r] = mnew;
#pragma unroll
      for (int c = 0; c < 4; ++c) o[c][r] *= corr;
      P[r + 8 * g][nn] = (_Float16)p0;
      P[r + 8 * g][16 + nn] = (_Float16)p1;
    }
    __syncthreads();
    v16h pa = load_a16(&P[0][0], 32, 0, 0); // C-layout -> A-layout via LDS
#pragma unroll
    for (int c = 0; c < 4; ++c) {
      v16h bv = load_b16(Vb, VPAD, c * 16, kt); // B[key, d-chunk] from transposed V
      o[c] = wmma_f16(pa, bv, o[c]);
    }
    __syncthreads();
  }

#pragma unroll
  for (int r = 0; r < 8; ++r) {
    int qrow = qt * 16 + r + 8 * g;
    float inv = lrow[r] > 0.f ? 1.0f / lrow[r] : 0.f;
    if (qrow < V_) {
#pragma unroll
      for (int c = 0; c < 4; ++c) {
        int col = h * DH + c * 16 + nn;
        o16[((size_t)b * V_ + qrow) * C_ + col] = (_Float16)(o[c][r] * inv);
      }
    }
  }
}

// ---------------- host orchestration ----------------
static inline dim3 ggrid(int n, int b) { return dim3((n + b - 1) / b); }

extern "C" void kernel_launch(void* const* d_in, const int* in_sizes, int n_in,
                              void* d_out, int out_size, void* d_ws, size_t ws_size,
                              hipStream_t stream) {
  const float* tokens  = (const float*)d_in[0];
  const float* centers = (const float*)d_in[1];
  const unsigned char* mask = (const unsigned char*)d_in[2];
  const float* pos_w1 = (const float*)d_in[3];
  const float* pos_b1 = (const float*)d_in[4];
  const float* pos_w2 = (const float*)d_in[5];
  const float* pos_b2 = (const float*)d_in[6];
  const float* ln1_w = (const float*)d_in[7];
  const float* ln1_b = (const float*)d_in[8];
  const float* qkv_w = (const float*)d_in[9];
  const float* qkv_b = (const float*)d_in[10];
  const float* proj_w = (const float*)d_in[11];
  const float* proj_b = (const float*)d_in[12];
  const float* ln2_w = (const float*)d_in[13];
  const float* ln2_b = (const float*)d_in[14];
  const float* fc1_w = (const float*)d_in[15];
  const float* fc1_b = (const float*)d_in[16];
  const float* fc2_w = (const float*)d_in[17];
  const float* fc2_b = (const float*)d_in[18];
  const float* lnf_w = (const float*)d_in[19];
  const float* lnf_b = (const float*)d_in[20];

  char* wp = (char*)d_ws;
  auto carve = [&](size_t bytes) -> char* {
    char* p = wp;
    wp += (bytes + 255) & ~(size_t)255;
    return p;
  };
  float*     xbuf  = (float*)carve((size_t)M_ * C_ * 4);
  float*     hbuf  = (float*)carve((size_t)M_ * C_ * 4);
  float*     posb  = (float*)carve((size_t)M_ * C_ * 4);
  _Float16*  hn16  = (_Float16*)carve((size_t)M_ * C_ * 2);
  _Float16*  t16   = (_Float16*)carve((size_t)M_ * 4 * C_ * 2);
  _Float16*  o16   = (_Float16*)carve((size_t)M_ * C_ * 2);
  _Float16*  q16   = (_Float16*)carve((size_t)B_ * H_ * VPAD * DH * 2);
  _Float16*  k16   = (_Float16*)carve((size_t)B_ * H_ * VPAD * DH * 2);
  _Float16*  vT    = (_Float16*)carve((size_t)B_ * H_ * VPAD * DH * 2);
  float*     ctrb  = (float*)carve((size_t)M_ * 3 * 4);
  int*       order = (int*)carve((size_t)M_ * 4);
  uint32_t*  mb    = (uint32_t*)carve((size_t)(M_ + 64) * MW * 4);
  _Float16*  qkvT  = (_Float16*)carve((size_t)DEPTH_ * 3 * C_ * C_ * 2);
  _Float16*  projT = (_Float16*)carve((size_t)DEPTH_ * C_ * C_ * 2);
  _Float16*  fc1T  = (_Float16*)carve((size_t)DEPTH_ * 4 * C_ * C_ * 2);
  _Float16*  fc2T  = (_Float16*)carve((size_t)DEPTH_ * 4 * C_ * C_ * 2);
  _Float16*  pw2T  = (_Float16*)carve((size_t)C_ * C_ * 2);

  // ---- pack / prep ----
  order_kernel<<<1, 32, 0, stream>>>(mask, order);
  gather_tokens<<<ggrid(M_ * C_, 256), 256, 0, stream>>>(tokens, order, xbuf);
  gather_ctr<<<ggrid(M_ * 3, 256), 256, 0, stream>>>(centers, order, ctrb);
  maskbits_kernel<<<ggrid(B_ * V_ * 26, 256), 256, 0, stream>>>(ctrb, mb);
  padv_kernel<<<ggrid(B_ * H_ * DH * (VPAD - V_), 256), 256, 0, stream>>>(vT);

  // ---- weight conversion (f32 [K][N] -> f16 [N][K]) ----
  convT_kernel<<<ggrid(C_ * C_, 256), 256, 0, stream>>>(pos_w2, pw2T, C_, C_, C_ * C_);
  for (int l = 0; l < DEPTH_; ++l) {
    convT_kernel<<<ggrid(C_ * 3 * C_, 256), 256, 0, stream>>>(
        qkv_w + (size_t)l * C_ * 3 * C_, qkvT + (size_t)l * 3 * C_ * C_, C_, 3 * C_, C_ * 3 * C_);
    convT_kernel<<<ggrid(C_ * C_, 256), 256, 0, stream>>>(
        proj_w + (size_t)l * C_ * C_, projT + (size_t)l * C_ * C_, C_, C_, C_ * C_);
    convT_kernel<<<ggrid(C_ * 4 * C_, 256), 256, 0, stream>>>(
        fc1_w + (size_t)l * C_ * 4 * C_, fc1T + (size_t)l * 4 * C_ * C_, C_, 4 * C_, C_ * 4 * C_);
    convT_kernel<<<ggrid(4 * C_ * C_, 256), 256, 0, stream>>>(
        fc2_w + (size_t)l * 4 * C_ * C_, fc2T + (size_t)l * 4 * C_ * C_, 4 * C_, C_, 4 * C_ * C_);
  }

  // ---- positional embedding MLP ----
  posmlp1_kernel<<<ggrid(M_ * C_, 256), 256, 0, stream>>>(ctrb, pos_w1, pos_b1, hn16);
  gemm_wmma<0><<<dim3(M_ / 16, C_ / 64), 32, 0, stream>>>(
      hn16, pw2T, pos_b2, C_, C_, posb, nullptr, nullptr, nullptr, nullptr, nullptr);

  // ---- transformer blocks ----
  for (int l = 0; l < DEPTH_; ++l) {
    // h = x + pos ; hn = LN1(h)
    ln_kernel<0><<<M_, 128, 0, stream>>>(xbuf, posb, ln1_w + l * C_, ln1_b + l * C_,
                                         hbuf, hn16, nullptr);
    // qkv
    gemm_wmma<3><<<dim3(M_ / 16, (3 * C_) / 64), 32, 0, stream>>>(
        hn16, qkvT + (size_t)l * 3 * C_ * C_, qkv_b + (size_t)l * 3 * C_,
        C_, 3 * C_, nullptr, nullptr, nullptr, q16, k16, vT);
    // masked attention
    attn_kernel<<<B_ * H_ * (VPAD / 16), 32, 0, stream>>>(q16, k16, vT, mb, o16);
    // h = h + proj(o)
    gemm_wmma<1><<<dim3(M_ / 16, C_ / 64), 32, 0, stream>>>(
        o16, projT + (size_t)l * C_ * C_, proj_b + (size_t)l * C_,
        C_, C_, hbuf, hbuf, nullptr, nullptr, nullptr, nullptr);
    // hn = LN2(h)
    ln_kernel<1><<<M_, 128, 0, stream>>>(hbuf, nullptr, ln2_w + l * C_, ln2_b + l * C_,
                                         nullptr, hn16, nullptr);
    // t = gelu(hn @ fc1)
    gemm_wmma<2><<<dim3(M_ / 16, (4 * C_) / 64), 32, 0, stream>>>(
        hn16, fc1T + (size_t)l * 4 * C_ * C_, fc1_b + (size_t)l * 4 * C_,
        C_, 4 * C_, nullptr, nullptr, t16, nullptr, nullptr, nullptr);
    // x = h + t @ fc2
    gemm_wmma<1><<<dim3(M_ / 16, C_ / 64), 32, 0, stream>>>(
        t16, fc2T + (size_t)l * 4 * C_ * C_, fc2_b + (size_t)l * C_,
        4 * C_, C_, xbuf, hbuf, nullptr, nullptr, nullptr, nullptr);
  }

  // ---- final LN -> d_out (f32) ----
  ln_kernel<2><<<M_, 128, 0, stream>>>(xbuf, nullptr, lnf_w, lnf_b,
                                       nullptr, nullptr, (float*)d_out);
  (void)in_sizes; (void)n_in; (void)out_size; (void)ws_size;
}
